// BigBirdProteinModel_77644418777790
// MI455X (gfx1250) — compile-verified, matrix operand order checked by reading, not run on hardware
//
#include <hip/hip_runtime.h>
#include <hip/hip_bf16.h>

// ---------------------------------------------------------------------------
// BigBird protein model forward for MI455X (gfx1250, wave32, WMMA).
// GEMMs + attention matmuls: v_wmma_f32_16x16x32_bf16.
// GEMM tiles: per-lane async global->LDS b128 (ASYNCcnt, double buffered).
// Attention K/V tiles: Tensor Data Mover DMA (TENSORcnt, double buffered,
//   hardware row padding -> bank-conflict-free [64][72] LDS layout).
// ---------------------------------------------------------------------------

#define SEQ     4096
#define INDIM   1280
#define DMODEL  768
#define NHEAD   12
#define DHEAD   64
#define BSZ     64
#define NBLK    64
#define NMID    62
#define CBLK    8
#define FFDIM   3072
#define NLAYER  12
#define CLSH    512
#define NOUT    2000
#define ATT_SCALE 0.125f
#define NEGINF  -1000000000.0f

typedef __attribute__((ext_vector_type(16))) __bf16 v16bf;
typedef __attribute__((ext_vector_type(8)))  float  v8f;
typedef __attribute__((ext_vector_type(4)))  unsigned v4u;
typedef __attribute__((ext_vector_type(8)))  unsigned v8u;

// ---------------- helpers ----------------

__device__ __forceinline__ unsigned short f2bf(float f) {
  unsigned u = __builtin_bit_cast(unsigned, f);
  u += 0x7FFFu + ((u >> 16) & 1u);
  return (unsigned short)(u >> 16);
}
__device__ __forceinline__ __bf16 bfraw(unsigned short h) {
  return __builtin_bit_cast(__bf16, h);
}
// CDNA5 16-bit A/B fragment K mapping (ISA 7.12.2)
__device__ __forceinline__ int kmap16(int j, int hi) {
  return ((j < 8) ? j : (8 + j)) + (hi << 3);
}
__device__ __forceinline__ float gelu_tanh(float x) {
  return 0.5f * x * (1.0f + tanhf(0.7978845608f * (x + 0.044715f * x * x * x)));
}

// per-lane async global->LDS copy (16B), tracked by ASYNCcnt
__device__ __forceinline__ void async_copy_b128(void* lds_dst, const void* gsrc) {
  unsigned l = (unsigned)(unsigned long long)lds_dst;   // LDS aperture: low 32 bits
  asm volatile("global_load_async_to_lds_b128 %0, %1, off"
               :: "v"(l), "v"(gsrc) : "memory");
}
#define S_WAIT_ASYNCCNT(n)  asm volatile("s_wait_asynccnt " #n ::: "memory")
#define S_WAIT_TENSORCNT(n) asm volatile("s_wait_tensorcnt " #n ::: "memory")

// Tensor Data Mover: DMA one 64x64 bf16 tile (row stride = stride elems) into
// LDS with 16B padding after each 128B row (pad_interval=32 DW, pad_amount=4 DW)
// => LDS layout matches unsigned short[64][72]. D# per ISA 8.3/8.4.
__device__ __forceinline__ void tdm_load_tile_64x64_bf16(
    unsigned lds_addr, const unsigned short* gsrc, unsigned stride_elems)
{
  unsigned long long ga = (unsigned long long)gsrc;
  v4u g0;
  g0[0] = 1u;                                            // count=1 user D#
  g0[1] = lds_addr;                                      // lds_addr (bytes)
  g0[2] = (unsigned)ga;                                  // global addr [31:0]
  g0[3] = (unsigned)((ga >> 32) & 0x1FFFFFFu) | (2u << 30); // addr[56:32]|type=2
  v8u g1;
  g1[0] = (1u << 16) | (1u << 20) | (4u << 22) | (3u << 25);
  //        data_size=2B  pad_en     pad_int=32DW  pad_amt=4DW
  g1[1] = 64u << 16;                 // tensor_dim0 = 64 (low16 @ bits 63:48)
  g1[2] = 64u << 16;                 // dim0 hi16=0 | tensor_dim1=64 low16
  g1[3] = 64u << 16;                 // dim1 hi16=0 | tile_dim0=64
  g1[4] = 64u;                       // tile_dim1=64 | tile_dim2=0
  g1[5] = stride_elems;              // tensor_dim0_stride low32
  g1[6] = 0u;                        // stride hi16 | dim1_stride low16
  g1[7] = 0u;                        // dim1_stride hi32
  asm volatile("tensor_load_to_lds %0, %1" :: "s"(g0), "s"(g1) : "memory");
}

// ---------------- elementwise kernels ----------------

__global__ void f32_to_bf16(const float* __restrict__ s,
                            unsigned short* __restrict__ d, long n4) {
  long i = (long)blockIdx.x * blockDim.x + threadIdx.x;
  long stride = (long)gridDim.x * blockDim.x;
  for (; i < n4; i += stride) {
    float4 v = ((const float4*)s)[i];
    ushort4 o;
    o.x = f2bf(v.x); o.y = f2bf(v.y); o.z = f2bf(v.z); o.w = f2bf(v.w);
    ((ushort4*)d)[i] = o;
  }
}

__global__ void mask_kernel(const int* __restrict__ am,
                            float* __restrict__ maskadd, int n) {
  int i = blockIdx.x * blockDim.x + threadIdx.x;
  if (i < n) maskadd[i] = (am[i] > 0) ? 0.f : NEGINF;
}

// ---------------- tiled bf16 WMMA GEMM, async double-buffered --------------
// EPI: 0 = f32 out, 1 = bf16 out, 2 = bf16 out + GELU(tanh)

template <int EPI>
__global__ __launch_bounds__(256) void gemm_bf16_wmma(
    const unsigned short* __restrict__ A, const unsigned short* __restrict__ B,
    const float* __restrict__ bias, void* __restrict__ C, int M, int N, int K)
{
  __shared__ __align__(16) unsigned short As[2][64][40];
  __shared__ __align__(16) unsigned short Bs[2][32][72];
  const int tid  = threadIdx.x;
  const int lane = tid & 31, wave = tid >> 5;
  const int ll = lane & 15,  hi = lane >> 4;
  const int m0 = blockIdx.y * 64, n0 = blockIdx.x * 64;
  const int wm = (wave & 3) << 4;
  const int wn = (wave >> 2) << 4;
  const int ra = tid >> 2, ca = (tid & 3) * 8;
  const int rb = tid >> 3, cb = (tid & 7) * 8;
  const int nk = K >> 5;

  v8f acc0 = {0.f,0.f,0.f,0.f,0.f,0.f,0.f,0.f};
  v8f acc1 = acc0;

  async_copy_b128(&As[0][ra][ca], A + (size_t)(m0 + ra) * K + ca);
  async_copy_b128(&Bs[0][rb][cb], B + (size_t)rb * N + n0 + cb);

  for (int it = 0; it < nk; ++it) {
    const int buf = it & 1;
    if (it + 1 < nk) {
      const int k1 = (it + 1) << 5;
      async_copy_b128(&As[buf ^ 1][ra][ca], A + (size_t)(m0 + ra) * K + k1 + ca);
      async_copy_b128(&Bs[buf ^ 1][rb][cb], B + (size_t)(k1 + rb) * N + n0 + cb);
      S_WAIT_ASYNCCNT(2);
    } else {
      S_WAIT_ASYNCCNT(0);
    }
    __syncthreads();
    v16bf af, bf0, bf1;
    #pragma unroll
    for (int j = 0; j < 16; ++j) {
      int kk = kmap16(j, hi);
      af[j]  = bfraw(As[buf][wm + ll][kk]);
      bf0[j] = bfraw(Bs[buf][kk][wn + ll]);
      bf1[j] = bfraw(Bs[buf][kk][wn + 32 + ll]);
    }
    acc0 = __builtin_amdgcn_wmma_f32_16x16x32_bf16(false, af, false, bf0,
                                                   (short)0, acc0, false, false);
    acc1 = __builtin_amdgcn_wmma_f32_16x16x32_bf16(false, af, false, bf1,
                                                   (short)0, acc1, false, false);
    __syncthreads();
  }
  #pragma unroll
  for (int r = 0; r < 8; ++r) {
    int m  = m0 + wm + r + (hi << 3);
    int na = n0 + wn + ll, nb = na + 32;
    float v0 = acc0[r] + bias[na];
    float v1 = acc1[r] + bias[nb];
    if (EPI == 2) { v0 = gelu_tanh(v0); v1 = gelu_tanh(v1); }
    if (EPI == 0) {
      ((float*)C)[(size_t)m * N + na] = v0;
      ((float*)C)[(size_t)m * N + nb] = v1;
    } else {
      ((unsigned short*)C)[(size_t)m * N + na] = f2bf(v0);
      ((unsigned short*)C)[(size_t)m * N + nb] = f2bf(v1);
    }
  }
}

// ---------------- LayerNorm kernels (emit f32 + bf16) ----------------------

__device__ __forceinline__ float block_reduce_sum256(float v, float* red) {
  int tid = threadIdx.x;
  red[tid] = v; __syncthreads();
  #pragma unroll
  for (int s = 128; s > 0; s >>= 1) {
    if (tid < s) red[tid] += red[tid + s];
    __syncthreads();
  }
  float r = red[0]; __syncthreads();
  return r;
}

__global__ __launch_bounds__(256) void embed_ln_kernel(
    const float* __restrict__ hp, const float* __restrict__ pos,
    const float* __restrict__ tok, const float* __restrict__ g,
    const float* __restrict__ b, float* __restrict__ out,
    unsigned short* __restrict__ outb)
{
  __shared__ float red[256];
  const int s = blockIdx.x, tid = threadIdx.x;
  float v[3];
  #pragma unroll
  for (int i = 0; i < 3; ++i) {
    int d = tid + i * 256;
    v[i] = hp[(size_t)s * DMODEL + d] + pos[(size_t)s * DMODEL + d] + tok[d];
  }
  float mu  = block_reduce_sum256(v[0] + v[1] + v[2], red) * (1.0f / DMODEL);
  float var = block_reduce_sum256((v[0]-mu)*(v[0]-mu) + (v[1]-mu)*(v[1]-mu) +
                                  (v[2]-mu)*(v[2]-mu), red) * (1.0f / DMODEL);
  float rs = rsqrtf(var + 1e-12f);
  #pragma unroll
  for (int i = 0; i < 3; ++i) {
    int d = tid + i * 256;
    float o = (v[i] - mu) * rs * g[d] + b[d];
    out[(size_t)s * DMODEL + d]  = o;
    outb[(size_t)s * DMODEL + d] = f2bf(o);
  }
}

__global__ __launch_bounds__(256) void add_ln_kernel(
    float* __restrict__ h, const float* __restrict__ a,
    const float* __restrict__ g, const float* __restrict__ b,
    unsigned short* __restrict__ hb)
{
  __shared__ float red[256];
  const int s = blockIdx.x, tid = threadIdx.x;
  float v[3];
  #pragma unroll
  for (int i = 0; i < 3; ++i) {
    int d = tid + i * 256;
    v[i] = h[(size_t)s * DMODEL + d] + a[(size_t)s * DMODEL + d];
  }
  float mu  = block_reduce_sum256(v[0] + v[1] + v[2], red) * (1.0f / DMODEL);
  float var = block_reduce_sum256((v[0]-mu)*(v[0]-mu) + (v[1]-mu)*(v[1]-mu) +
                                  (v[2]-mu)*(v[2]-mu), red) * (1.0f / DMODEL);
  float rs = rsqrtf(var + 1e-12f);
  #pragma unroll
  for (int i = 0; i < 3; ++i) {
    int d = tid + i * 256;
    float o = (v[i] - mu) * rs * g[d] + b[d];
    h[(size_t)s * DMODEL + d]  = o;
    hb[(size_t)s * DMODEL + d] = f2bf(o);
  }
}

// ---------------- BigBird block plan ----------------

__global__ void plan_kernel(int* __restrict__ idx, int* __restrict__ valid) {
  if (threadIdx.x != 0) return;
  const int ib = blockIdx.x, h = blockIdx.y, iq = ib + 1;
  int v[CBLK];
  v[0] = 0; v[1] = iq - 1; v[2] = iq; v[3] = iq + 1; v[4] = NBLK - 1;
  unsigned seed = 0x9E3779B9u ^ (unsigned)(h * 7919 + ib * 104729);
  for (int r = 5; r < CBLK; ++r) {
    int cand = 1;
    for (;;) {
      seed = seed * 1664525u + 1013904223u;
      cand = 1 + (int)((seed >> 10) % (unsigned)(NBLK - 2));
      bool clash = false;
      for (int a = 0; a < r; ++a) clash = clash || (v[a] == cand);
      if (!clash) break;
    }
    v[r] = cand;
  }
  const int base = (h * NMID + ib) * CBLK;
  for (int a = 0; a < CBLK; ++a) {
    int ok = 1;
    for (int p = 0; p < a; ++p) if (v[a] == v[p]) ok = 0;
    idx[base + a]   = v[a];
    valid[base + a] = ok;
  }
}

// ---------------- fused middle-block sparse attention ----------------------
// grid (62, H); TDM double-buffered K/V tiles; 64x512 score tile in LDS.

__global__ __launch_bounds__(256) void attn_middle_kernel(
    const unsigned short* __restrict__ qb, const unsigned short* __restrict__ kb,
    const unsigned short* __restrict__ vb, const int* __restrict__ idxp,
    const int* __restrict__ validp, const float* __restrict__ maskadd,
    unsigned short* __restrict__ ob)
{
  __shared__ __align__(16) unsigned short Qs[64][72];
  __shared__ __align__(16) unsigned short KVs[2][64][72];
  __shared__ float Sc[64][CBLK * 64 + 4];
  __shared__ float mk[64];
  __shared__ int   kidx[CBLK], kval[CBLK];
  __shared__ float rowmax[64], rowsum[64], red4[64][4];

  const int ib = blockIdx.x, h = blockIdx.y, iq = ib + 1;
  const int tid = threadIdx.x, lane = tid & 31, wave = tid >> 5;
  const int ll = lane & 15, hi = lane >> 4;
  const int wm = (wave & 3) << 4, wn = (wave >> 2) << 4;

  if (tid < CBLK) {
    kidx[tid] = idxp[(h * NMID + ib) * CBLK + tid];
    kval[tid] = validp[(h * NMID + ib) * CBLK + tid];
  }
  {
    const unsigned short* qs = qb + (size_t)(iq * BSZ) * DMODEL + h * DHEAD;
    for (int t = tid; t < 512; t += 256) {
      int r = t >> 3, c = (t & 7) * 8;
      *(uint4*)&Qs[r][c] = *(const uint4*)(qs + (size_t)r * DMODEL + c);
    }
  }
  __syncthreads();

  // wave 0 drives the TDM pipeline
  auto issue_tdm = [&](const unsigned short* base, int blk,
                       unsigned short (*dst)[72]) {
    tdm_load_tile_64x64_bf16((unsigned)(unsigned long long)&dst[0][0],
                             base + (size_t)(blk * BSZ) * DMODEL + h * DHEAD,
                             (unsigned)DMODEL);
  };

  if (wave == 0)
    issue_tdm(kb, __builtin_amdgcn_readfirstlane(kidx[0]), KVs[0]);

  // ---- S = Q K^T per gathered block ----
  for (int ci = 0; ci < CBLK; ++ci) {
    if (tid < 64)
      mk[tid] = kval[ci] ? maskadd[kidx[ci] * BSZ + tid] : NEGINF;
    if (wave == 0) {
      if (ci + 1 < CBLK) {
        issue_tdm(kb, __builtin_amdgcn_readfirstlane(kidx[ci + 1]),
                  KVs[(ci + 1) & 1]);
        S_WAIT_TENSORCNT(1);
      } else {
        S_WAIT_TENSORCNT(0);
      }
    }
    __syncthreads();
    v8f s0 = {0.f,0.f,0.f,0.f,0.f,0.f,0.f,0.f};
    v8f s1 = s0;
    const int buf = ci & 1;
    #pragma unroll
    for (int kd = 0; kd < 64; kd += 32) {
      v16bf af, b0, b1;
      #pragma unroll
      for (int j = 0; j < 16; ++j) {
        int kk = kd + kmap16(j, hi);
        af[j] = bfraw(Qs[wm + ll][kk]);
        b0[j] = bfraw(KVs[buf][wn + ll][kk]);
        b1[j] = bfraw(KVs[buf][wn + 32 + ll][kk]);
      }
      s0 = __builtin_amdgcn_wmma_f32_16x16x32_bf16(false, af, false, b0,
                                                   (short)0, s0, false, false);
      s1 = __builtin_amdgcn_wmma_f32_16x16x32_bf16(false, af, false, b1,
                                                   (short)0, s1, false, false);
    }
    #pragma unroll
    for (int r = 0; r < 8; ++r) {
      int m = wm + r + (hi << 3);
      int na = wn + ll, nb = na + 32;
      Sc[m][ci * 64 + na] = s0[r] * ATT_SCALE + mk[na];
      Sc[m][ci * 64 + nb] = s1[r] * ATT_SCALE + mk[nb];
    }
    __syncthreads();
  }

  // stage V tile 0 (overlaps softmax)
  if (wave == 0)
    issue_tdm(vb, __builtin_amdgcn_readfirstlane(kidx[0]), KVs[0]);

  // ---- row softmax over 512 cols (4 threads per row) ----
  {
    const int row = tid >> 2, part = tid & 3;
    float lm = -3.0e38f;
    for (int j = part * 128; j < part * 128 + 128; ++j) lm = fmaxf(lm, Sc[row][j]);
    red4[row][part] = lm; __syncthreads();
    if (part == 0)
      rowmax[row] = fmaxf(fmaxf(red4[row][0], red4[row][1]),
                          fmaxf(red4[row][2], red4[row][3]));
    __syncthreads();
    const float rm = rowmax[row];
    float ls = 0.f;
    for (int j = part * 128; j < part * 128 + 128; ++j) {
      float p = __expf(Sc[row][j] - rm);
      Sc[row][j] = p;
      ls += p;
    }
    red4[row][part] = ls; __syncthreads();
    if (part == 0)
      rowsum[row] = red4[row][0] + red4[row][1] + red4[row][2] + red4[row][3];
    __syncthreads();
  }

  // ---- O = P V per gathered block ----
  v8f o0 = {0.f,0.f,0.f,0.f,0.f,0.f,0.f,0.f};
  v8f o1 = o0;
  for (int ci = 0; ci < CBLK; ++ci) {
    if (wave == 0) {
      if (ci + 1 < CBLK) {
        issue_tdm(vb, __builtin_amdgcn_readfirstlane(kidx[ci + 1]),
                  KVs[(ci + 1) & 1]);
        S_WAIT_TENSORCNT(1);
      } else {
        S_WAIT_TENSORCNT(0);
      }
    }
    __syncthreads();
    const int buf = ci & 1;
    #pragma unroll
    for (int kd = 0; kd < 64; kd += 32) {
      v16bf af, b0, b1;
      #pragma unroll
      for (int j = 0; j < 16; ++j) {
        int kl = kd + kmap16(j, hi);
        af[j] = bfraw(f2bf(Sc[wm + ll][ci * 64 + kl]));
        b0[j] = bfraw(KVs[buf][kl][wn + ll]);
        b1[j] = bfraw(KVs[buf][kl][wn + 32 + ll]);
      }
      o0 = __builtin_amdgcn_wmma_f32_16x16x32_bf16(false, af, false, b0,
                                                   (short)0, o0, false, false);
      o1 = __builtin_amdgcn_wmma_f32_16x16x32_bf16(false, af, false, b1,
                                                   (short)0, o1, false, false);
    }
    __syncthreads();
  }
  #pragma unroll
  for (int r = 0; r < 8; ++r) {
    int m = wm + r + (hi << 3);
    float inv = 1.0f / (rowsum[m] + 1e-20f);
    size_t rowoff = (size_t)(iq * BSZ + m) * DMODEL + h * DHEAD;
    ob[rowoff + wn + ll]      = f2bf(o0[r] * inv);
    ob[rowoff + wn + 32 + ll] = f2bf(o1[r] * inv);
  }
}

// ---------------- fused global attention (q-blocks 0 and 63) ---------------
// grid (2, H); flash-style online softmax; TDM double-buffered K/V tiles.

__global__ __launch_bounds__(256) void attn_global_kernel(
    const unsigned short* __restrict__ qb, const unsigned short* __restrict__ kb,
    const unsigned short* __restrict__ vb, const float* __restrict__ maskadd,
    unsigned short* __restrict__ ob)
{
  __shared__ __align__(16) unsigned short Qs[64][72];
  __shared__ __align__(16) unsigned short Ks[2][64][72];
  __shared__ __align__(16) unsigned short Vs[2][64][72];
  __shared__ float Sc[64][68];
  __shared__ float mk[64];
  __shared__ float mrow[64], lrow[64], srow[64], red4[64][4];

  const int gq = blockIdx.x, h = blockIdx.y;
  const int q0 = (gq == 0) ? 0 : (SEQ - BSZ);
  const int tid = threadIdx.x, lane = tid & 31, wave = tid >> 5;
  const int ll = lane & 15, hi = lane >> 4;
  const int wm = (wave & 3) << 4, wn = (wave >> 2) << 4;

  if (tid < 64) { mrow[tid] = -3.0e38f; lrow[tid] = 0.f; }
  {
    const unsigned short* qs = qb + (size_t)q0 * DMODEL + h * DHEAD;
    for (int t = tid; t < 512; t += 256) {
      int r = t >> 3, c = (t & 7) * 8;
      *(uint4*)&Qs[r][c] = *(const uint4*)(qs + (size_t)r * DMODEL + c);
    }
  }
  __syncthreads();

  if (wave == 0) {
    tdm_load_tile_64x64_bf16((unsigned)(unsigned long long)&Ks[0][0][0],
                             kb + h * DHEAD, (unsigned)DMODEL);
    tdm_load_tile_64x64_bf16((unsigned)(unsigned long long)&Vs[0][0][0],
                             vb + h * DHEAD, (unsigned)DMODEL);
  }

  v8f o0 = {0.f,0.f,0.f,0.f,0.f,0.f,0.f,0.f};
  v8f o1 = o0;

  for (int kbi = 0; kbi < NBLK; ++kbi) {
    if (tid < 64) mk[tid] = maskadd[kbi * BSZ + tid];
    if (wave == 0) {
      if (kbi + 1 < NBLK) {
        const size_t nb0 = (size_t)((kbi + 1) * BSZ) * DMODEL + h * DHEAD;
        tdm_load_tile_64x64_bf16(
            (unsigned)(unsigned long long)&Ks[(kbi + 1) & 1][0][0],
            kb + nb0, (unsigned)DMODEL);
        tdm_load_tile_64x64_bf16(
            (unsigned)(unsigned long long)&Vs[(kbi + 1) & 1][0][0],
            vb + nb0, (unsigned)DMODEL);
        S_WAIT_TENSORCNT(2);
      } else {
        S_WAIT_TENSORCNT(0);
      }
    }
    __syncthreads();
    const int buf = kbi & 1;

    v8f s0 = {0.f,0.f,0.f,0.f,0.f,0.f,0.f,0.f};
    v8f s1 = s0;
    #pragma unroll
    for (int kd = 0; kd < 64; kd += 32) {
      v16bf af, b0, b1;
      #pragma unroll
      for (int j = 0; j < 16; ++j) {
        int kk = kd + kmap16(j, hi);
        af[j] = bfraw(Qs[wm + ll][kk]);
        b0[j] = bfraw(Ks[buf][wn + ll][kk]);
        b1[j] = bfraw(Ks[buf][wn + 32 + ll][kk]);
      }
      s0 = __builtin_amdgcn_wmma_f32_16x16x32_bf16(false, af, false, b0,
                                                   (short)0, s0, false, false);
      s1 = __builtin_amdgcn_wmma_f32_16x16x32_bf16(false, af, false, b1,
                                                   (short)0, s1, false, false);
    }
    #pragma unroll
    for (int r = 0; r < 8; ++r) {
      int m = wm + r + (hi << 3);
      int na = wn + ll, nb = na + 32;
      Sc[m][na] = s0[r] * ATT_SCALE + mk[na];
      Sc[m][nb] = s1[r] * ATT_SCALE + mk[nb];
    }
    __syncthreads();

    // online softmax update (4 threads per row)
    {
      const int row = tid >> 2, part = tid & 3;
      float lm = -3.0e38f;
      for (int j = part * 16; j < part * 16 + 16; ++j) lm = fmaxf(lm, Sc[row][j]);
      red4[row][part] = lm; __syncthreads();
      if (part == 0) {
        float nm = fmaxf(fmaxf(red4[row][0], red4[row][1]),
                         fmaxf(red4[row][2], red4[row][3]));
        nm = fmaxf(nm, mrow[row]);
        float sc = __expf(mrow[row] - nm);
        srow[row] = sc;
        lrow[row] *= sc;
        mrow[row] = nm;
      }
      __syncthreads();
      const float nm = mrow[row];
      float ls = 0.f;
      for (int j = part * 16; j < part * 16 + 16; ++j) {
        float p = __expf(Sc[row][j] - nm);
        Sc[row][j] = p;
        ls += p;
      }
      red4[row][part] = ls; __syncthreads();
      if (part == 0)
        lrow[row] += red4[row][0] + red4[row][1] + red4[row][2] + red4[row][3];
      __syncthreads();
    }

    #pragma unroll
    for (int r = 0; r < 8; ++r) {
      float sc = srow[wm + r + (hi << 3)];
      o0[r] *= sc; o1[r] *= sc;
    }
    #pragma unroll
    for (int kd = 0; kd < 64; kd += 32) {
      v16bf af, b0, b1;
      #pragma unroll
      for (int j = 0; j < 16; ++j) {
        int kl = kd + kmap16(j, hi);
        af[j] = bfraw(f2bf(Sc[wm + ll][kl]));
        b0[j] = bfraw(Vs[buf][kl][wn + ll]);
        b1[j] = bfraw(Vs[buf][kl][wn + 32 + ll]);
      }
      o0 = __builtin_amdgcn_wmma_f32_16x16x32_bf16(false, af, false, b0,
                                                   (short)0, o0, false, false);
      o1 = __builtin_amdgcn_wmma_f32_16x16x32_bf16(false, af, false, b1,
                                                   (short)0, o1, false, false);
    }
    __syncthreads();
  }

  #pragma unroll
  for (int r = 0; r < 8; ++r) {
    int m = wm + r + (hi << 3);
    float inv = 1.0f / (lrow[m] + 1e-20f);
    size_t rowoff = (size_t)(q0 + m) * DMODEL + h * DHEAD;
    ob[rowoff + wn + ll]      = f2bf(o0[r] * inv);
    ob[rowoff + wn + 32 + ll] = f2bf(o1[r] * inv);
  }
}

// ---------------- classifier head ----------------

__global__ __launch_bounds__(64) void fc_kernel(
    const float* __restrict__ x, const float* __restrict__ W,
    const float* __restrict__ b, float* __restrict__ out,
    int K, int N, int relu)
{
  __shared__ float red[64];
  const int j = blockIdx.x;
  float s = 0.f;
  for (int k = threadIdx.x; k < K; k += 64) s += x[k] * W[(size_t)k * N + j];
  red[threadIdx.x] = s; __syncthreads();
  #pragma unroll
  for (int t = 32; t > 0; t >>= 1) {
    if (threadIdx.x < t) red[threadIdx.x] += red[threadIdx.x + t];
    __syncthreads();
  }
  if (threadIdx.x == 0) {
    float v = red[0] + b[j];
    out[j] = relu ? fmaxf(v, 0.f) : v;
  }
}

// ---------------------------------------------------------------------------
// host orchestration
// ---------------------------------------------------------------------------

extern "C" void kernel_launch(void* const* d_in, const int* in_sizes, int n_in,
                              void* d_out, int out_size, void* d_ws, size_t ws_size,
                              hipStream_t stream) {
  (void)in_sizes; (void)n_in; (void)out_size; (void)ws_size;

  const float* x    = (const float*)d_in[0];
  const int*   am   = (const int*)d_in[1];
  const float* Wp   = (const float*)d_in[2];
  const float* bp   = (const float*)d_in[3];
  const float* pos  = (const float*)d_in[4];
  const float* tok  = (const float*)d_in[5];
  const float* lng  = (const float*)d_in[6];
  const float* lnb  = (const float*)d_in[7];
  const float* Wq   = (const float*)d_in[8];
  const float* bq   = (const float*)d_in[9];
  const float* Wk   = (const float*)d_in[10];
  const float* bk   = (const float*)d_in[11];
  const float* Wv   = (const float*)d_in[12];
  const float* bv   = (const float*)d_in[13];
  const float* Wo   = (const float*)d_in[14];
  const float* bo   = (const float*)d_in[15];
  const float* ln1g = (const float*)d_in[16];
  const float* ln1b = (const float*)d_in[17];
  const float* W1   = (const float*)d_in[18];
  const float* b1f  = (const float*)d_in[19];
  const float* W2   = (const float*)d_in[20];
  const float* b2f  = (const float*)d_in[21];
  const float* ln2g = (const float*)d_in[22];
  const float* ln2b = (const float*)d_in[23];
  const float* Wc1  = (const float*)d_in[24];
  const float* bc1  = (const float*)d_in[25];
  const float* Wc2  = (const float*)d_in[26];
  const float* bc2  = (const float*)d_in[27];

  size_t off = 0;
  auto alloc = [&](size_t bytes) -> void* {
    void* p = (char*)d_ws + off;
    off += (bytes + 255) & ~(size_t)255;
    return p;
  };
  float*          h    = (float*)alloc((size_t)SEQ * DMODEL * 4);
  float*          tmp  = (float*)alloc((size_t)SEQ * DMODEL * 4);
  unsigned short* hb   = (unsigned short*)alloc((size_t)SEQ * DMODEL * 2);
  unsigned short* qbu  = (unsigned short*)alloc((size_t)SEQ * DMODEL * 2);
  unsigned short* kbu  = (unsigned short*)alloc((size_t)SEQ * DMODEL * 2);
  unsigned short* vbu  = (unsigned short*)alloc((size_t)SEQ * DMODEL * 2);
  unsigned short* obu  = (unsigned short*)alloc((size_t)SEQ * DMODEL * 2);
  unsigned short* ff1b = (unsigned short*)alloc((size_t)SEQ * FFDIM * 2);
  unsigned short* Wbuf = (unsigned short*)alloc((size_t)DMODEL * FFDIM * 2);
  unsigned short* xb   = (unsigned short*)alloc((size_t)SEQ * INDIM * 2);
  float*          mka  = (float*)alloc((size_t)SEQ * 4);
  int*            idx  = (int*)alloc((size_t)NHEAD * NMID * CBLK * 4);
  int*            vld  = (int*)alloc((size_t)NHEAD * NMID * CBLK * 4);
  float*          clsr = (float*)alloc((size_t)CLSH * 4);

  auto conv = [&](const float* s, unsigned short* d, long n) {
    f32_to_bf16<<<1024, 256, 0, stream>>>(s, d, n / 4);
  };

  plan_kernel<<<dim3(NMID, NHEAD), 64, 0, stream>>>(idx, vld);
  mask_kernel<<<SEQ / 256, 256, 0, stream>>>(am, mka, SEQ);

  conv(x, xb, (long)SEQ * INDIM);
  conv(Wp, Wbuf, (long)INDIM * DMODEL);
  gemm_bf16_wmma<0><<<dim3(DMODEL / 64, SEQ / 64), 256, 0, stream>>>(
      xb, Wbuf, bp, tmp, SEQ, DMODEL, INDIM);
  embed_ln_kernel<<<SEQ, 256, 0, stream>>>(tmp, pos, tok, lng, lnb, h, hb);

  for (int l = 0; l < NLAYER; ++l) {
    const size_t wsz = (size_t)DMODEL * DMODEL;
    const size_t fsz = (size_t)DMODEL * FFDIM;

    conv(Wq + (size_t)l * wsz, Wbuf, (long)wsz);
    gemm_bf16_wmma<1><<<dim3(DMODEL / 64, SEQ / 64), 256, 0, stream>>>(
        hb, Wbuf, bq + (size_t)l * DMODEL, qbu, SEQ, DMODEL, DMODEL);
    conv(Wk + (size_t)l * wsz, Wbuf, (long)wsz);
    gemm_bf16_wmma<1><<<dim3(DMODEL / 64, SEQ / 64), 256, 0, stream>>>(
        hb, Wbuf, bk + (size_t)l * DMODEL, kbu, SEQ, DMODEL, DMODEL);
    conv(Wv + (size_t)l * wsz, Wbuf, (long)wsz);
    gemm_bf16_wmma<1><<<dim3(DMODEL / 64, SEQ / 64), 256, 0, stream>>>(
        hb, Wbuf, bv + (size_t)l * DMODEL, vbu, SEQ, DMODEL, DMODEL);

    attn_middle_kernel<<<dim3(NMID, NHEAD), 256, 0, stream>>>(
        qbu, kbu, vbu, idx, vld, mka, obu);
    attn_global_kernel<<<dim3(2, NHEAD), 256, 0, stream>>>(
        qbu, kbu, vbu, mka, obu);

    conv(Wo + (size_t)l * wsz, Wbuf, (long)wsz);
    gemm_bf16_wmma<0><<<dim3(DMODEL / 64, SEQ / 64), 256, 0, stream>>>(
        obu, Wbuf, bo + (size_t)l * DMODEL, tmp, SEQ, DMODEL, DMODEL);
    add_ln_kernel<<<SEQ, 256, 0, stream>>>(h, tmp,
        ln1g + (size_t)l * DMODEL, ln1b + (size_t)l * DMODEL, hb);

    conv(W1 + (size_t)l * fsz, Wbuf, (long)fsz);
    gemm_bf16_wmma<2><<<dim3(FFDIM / 64, SEQ / 64), 256, 0, stream>>>(
        hb, Wbuf, b1f + (size_t)l * FFDIM, ff1b, SEQ, FFDIM, DMODEL);
    conv(W2 + (size_t)l * fsz, Wbuf, (long)fsz);
    gemm_bf16_wmma<0><<<dim3(DMODEL / 64, SEQ / 64), 256, 0, stream>>>(
        ff1b, Wbuf, b2f + (size_t)l * DMODEL, tmp, SEQ, DMODEL, FFDIM);
    add_ln_kernel<<<SEQ, 256, 0, stream>>>(h, tmp,
        ln2g + (size_t)l * DMODEL, ln2b + (size_t)l * DMODEL, hb);
  }

  fc_kernel<<<CLSH, 64, 0, stream>>>(h, Wc1, bc1, clsr, DMODEL, CLSH, 1);
  fc_kernel<<<NOUT, 64, 0, stream>>>(clsr, Wc2, bc2, (float*)d_out, CLSH, NOUT, 0);
}
